// PointRefinerGNN_33174327394812
// MI455X (gfx1250) — matrix operand order repaired and checked by reading, not compiled, not run on hardware
//
#include <hip/hip_runtime.h>
#include <stdint.h>

typedef __bf16 bf16_t;
typedef __attribute__((ext_vector_type(16))) __bf16 v16bf;
typedef __attribute__((ext_vector_type(8)))  __bf16 v8bf;
typedef __attribute__((ext_vector_type(8)))  float   v8f;
typedef int v4i __attribute__((vector_size(4 * sizeof(int))));

#if defined(__gfx1250__) && __has_builtin(__builtin_amdgcn_global_load_async_to_lds_b128) && __has_builtin(__builtin_amdgcn_s_wait_asynccnt)
#define HAVE_ASYNC 1
#define ASYNC_GPTR(p) ((__attribute__((address_space(1))) v4i*)(p))
#define ASYNC_LPTR(p) ((__attribute__((address_space(3))) v4i*)(p))
#else
#define HAVE_ASYNC 0
#endif

#define TILE_M 128
#define TILE_N 64
#define TILE_K 64   // two 16x16x32 K-substeps per staging round -> 8 WMMAs per barrier

// dinv[i] = rsqrt(1 + nnz(adj row i))   (self-loop included; adj diag is 0)
__global__ __launch_bounds__(256) void gcn_degree_kernel(const float* __restrict__ adj,
                                                         float* __restrict__ dinv, int n) {
  __shared__ float red[256];
  const int row = blockIdx.x;
  const float* r = adj + (long)row * n;
  float c = 0.0f;
  for (int j = threadIdx.x; j < n; j += 256) c += (r[j] != 0.0f) ? 1.0f : 0.0f;
  red[threadIdx.x] = c;
  __syncthreads();
  for (int s = 128; s > 0; s >>= 1) {
    if (threadIdx.x < s) red[threadIdx.x] += red[threadIdx.x + s];
    __syncthreads();
  }
  if (threadIdx.x == 0) dinv[row] = rsqrtf(red[0] + 1.0f);
}

// Generic tiled bf16 WMMA GEMM with fused GCN epilogues.
//   acc = Aop(A[MxK]) @ Bop(B[KxN])          (both row-major sources)
// a_mode: 0 = f32->bf16, 1 = bf16 passthrough (async-to-LDS path), 2 = f32 binarize (adjacency)
// b_mode: 0 = f32->bf16, 1 = bf16            (coalesced row loads + LDS transpose scatter)
// ep:     0 : outb[m,n] = bf16(dinv[m]*acc)
//         1 : outb[m,n] = bf16(relu(dinv[m]*(acc + selfB[m,n]) + bias[n]))
//         2 : outf[m,n] = resid[m,n] + alpha*(dinv[m]*(acc + selfB[m,n]) + bias[n])
__global__ __launch_bounds__(256) void gcn_wmma_gemm(
    const float* __restrict__ Af32, const bf16_t* __restrict__ Abf, int a_mode, int lda,
    const float* __restrict__ Bf32, const bf16_t* __restrict__ Bbf, int b_mode, int ldb,
    int M, int N, int K, int ep,
    const float* __restrict__ dinv, const float* __restrict__ bias,
    const bf16_t* __restrict__ selfB, int ldsb,
    const float* __restrict__ resid, const float* __restrict__ alpha_p,
    bf16_t* __restrict__ outb, float* __restrict__ outf, int ldo) {
  __shared__ alignas(32) bf16_t As[TILE_M * TILE_K];   // row-major [m][k], 16KB
  __shared__ alignas(32) bf16_t Bs[TILE_N * TILE_K];   // transposed [n][k], 8KB

  const int tid  = threadIdx.x;
  const int lane = tid & 31;
  const int wave = tid >> 5;   // 8 waves
  const int hf   = lane >> 4;  // lane half (ISA fragment layout)
  const int ln   = lane & 15;
  const int wm   = wave & 3;   // 4 waves along M (32 rows each)
  const int wn   = wave >> 2;  // 2 waves along N (32 cols each)
  const int m0   = blockIdx.y * TILE_M;
  const int n0   = blockIdx.x * TILE_N;

  v8f acc[2][2];
#pragma unroll
  for (int ti = 0; ti < 2; ++ti)
#pragma unroll
    for (int tj = 0; tj < 2; ++tj) {
      v8f z = {0.f, 0.f, 0.f, 0.f, 0.f, 0.f, 0.f, 0.f};
      acc[ti][tj] = z;
    }

  // A staging: thread -> 32 contiguous elements of one row (128B f32 / 64B bf16 requests)
  const int arow = tid >> 1;          // 0..127
  const int acol = (tid & 1) * 32;    // 0 or 32
  // B staging: thread -> 16 contiguous elements of one K-row (coalesced), scattered into Bs[n][k]
  const int bkr = tid >> 2;           // 0..63 (k row)
  const int bc0 = (tid & 3) * 16;     // col group

  for (int k0 = 0; k0 < K; k0 += TILE_K) {
    __syncthreads();  // previous iteration's fragment reads done before LDS overwrite
    // ---- stage A tile (128x64) into LDS ----
    {
      bf16_t* dst = &As[arow * TILE_K + acol];
      const long gb = (long)(m0 + arow) * lda + k0 + acol;
      if (a_mode == 0) {
        const float* s = Af32 + gb;
#pragma unroll
        for (int i = 0; i < 32; ++i) dst[i] = (bf16_t)s[i];
      } else if (a_mode == 2) {
        const float* s = Af32 + gb;
        if (k0 + TILE_K < K) __builtin_prefetch(s + TILE_K, 0, 1);  // global_prefetch next tile
#pragma unroll
        for (int i = 0; i < 32; ++i) dst[i] = (s[i] != 0.0f) ? (bf16_t)1.0f : (bf16_t)0.0f;
      } else {
        const bf16_t* s = Abf + gb;
#if HAVE_ASYNC
        __builtin_amdgcn_global_load_async_to_lds_b128(ASYNC_GPTR(s),      ASYNC_LPTR(dst),      0, 0);
        __builtin_amdgcn_global_load_async_to_lds_b128(ASYNC_GPTR(s +  8), ASYNC_LPTR(dst +  8), 0, 0);
        __builtin_amdgcn_global_load_async_to_lds_b128(ASYNC_GPTR(s + 16), ASYNC_LPTR(dst + 16), 0, 0);
        __builtin_amdgcn_global_load_async_to_lds_b128(ASYNC_GPTR(s + 24), ASYNC_LPTR(dst + 24), 0, 0);
#else
#pragma unroll
        for (int i = 0; i < 32; ++i) dst[i] = s[i];
#endif
      }
    }
    // ---- stage B tile (64x64): coalesced row loads, transpose-scatter into Bs[n][k] ----
    {
      if (b_mode == 0) {
        const float* s = Bf32 + (long)(k0 + bkr) * ldb + n0 + bc0;  // 64B contiguous
#pragma unroll
        for (int i = 0; i < 16; ++i) Bs[(bc0 + i) * TILE_K + bkr] = (bf16_t)s[i];
      } else {
        const bf16_t* s = Bbf + (long)(k0 + bkr) * ldb + n0 + bc0;  // 32B contiguous
        v8bf r0 = *(const v8bf*)s;
        v8bf r1 = *(const v8bf*)(s + 8);
#pragma unroll
        for (int i = 0; i < 8; ++i) {
          Bs[(bc0 + i) * TILE_K + bkr]     = r0[i];
          Bs[(bc0 + 8 + i) * TILE_K + bkr] = r1[i];
        }
      }
    }
#if HAVE_ASYNC
    if (a_mode == 1) __builtin_amdgcn_s_wait_asynccnt(0);
#endif
    __syncthreads();

    // ---- two K-substeps of WMMA per staging round ----
#pragma unroll
    for (int s = 0; s < 2; ++s) {
      const int kb = s * 32;
      v16bf afrag[2], bfrag[2];
#pragma unroll
      for (int ti = 0; ti < 2; ++ti) {
        const int row = wm * 32 + ti * 16 + ln;  // A: lane ln holds row M=ln
        const v8bf lo = *(const v8bf*)&As[row * TILE_K + kb + 8 * hf];       // K = 8h..8h+7
        const v8bf hi = *(const v8bf*)&As[row * TILE_K + kb + 16 + 8 * hf];  // K = 16+8h..+7
        v16bf a;
#pragma unroll
        for (int i = 0; i < 8; ++i) { a[i] = lo[i]; a[i + 8] = hi[i]; }
        afrag[ti] = a;
      }
#pragma unroll
      for (int tj = 0; tj < 2; ++tj) {
        const int nn = wn * 32 + tj * 16 + ln;  // B: lane ln holds column N=ln, K=8h..8h+7, 16+8h..+7
        const v8bf blo = *(const v8bf*)&Bs[nn * TILE_K + kb + 8 * hf];
        const v8bf bhi = *(const v8bf*)&Bs[nn * TILE_K + kb + 16 + 8 * hf];
        v16bf b;
#pragma unroll
        for (int i = 0; i < 8; ++i) { b[i] = blo[i]; b[i + 8] = bhi[i]; }
        bfrag[tj] = b;
      }
#pragma unroll
      for (int ti = 0; ti < 2; ++ti)
#pragma unroll
        for (int tj = 0; tj < 2; ++tj)
          acc[ti][tj] = __builtin_amdgcn_wmma_f32_16x16x32_bf16(
              false, afrag[ti], false, bfrag[tj], (short)0, acc[ti][tj], false, false);
    }
  }

  // ---- fused epilogue (C/D layout: element r -> M = r + 8*hf, N = ln) ----
  const float alpha = (ep == 2) ? *alpha_p : 0.0f;
#pragma unroll
  for (int ti = 0; ti < 2; ++ti) {
#pragma unroll
    for (int tj = 0; tj < 2; ++tj) {
      const int nn = n0 + wn * 32 + tj * 16 + ln;
#pragma unroll
      for (int r = 0; r < 8; ++r) {
        const int mm = m0 + wm * 32 + ti * 16 + r + 8 * hf;
        const float v = acc[ti][tj][r];
        if (ep == 0) {
          outb[(long)mm * ldo + nn] = (bf16_t)(dinv[mm] * v);
        } else if (ep == 1) {
          const float sum = v + (float)selfB[(long)mm * ldsb + nn];  // + self-loop
          float hv = dinv[mm] * sum + bias[nn];
          outb[(long)mm * ldo + nn] = (bf16_t)(hv > 0.0f ? hv : 0.0f);
        } else {
          const float sum = v + (float)selfB[(long)mm * ldsb + nn];
          const float yv = dinv[mm] * sum + bias[nn];
          outf[(long)mm * ldo + nn] = resid[(long)mm * ldo + nn] + alpha * yv;
        }
      }
    }
  }
}

extern "C" void kernel_launch(void* const* d_in, const int* in_sizes, int n_in,
                              void* d_out, int out_size, void* d_ws, size_t ws_size,
                              hipStream_t stream) {
  const int NN = 8192, IN = 1024, HID = 256;
  const float* x     = (const float*)d_in[0];
  const float* adj   = (const float*)d_in[1];
  const float* W1    = (const float*)d_in[2];
  const float* b1    = (const float*)d_in[3];
  const float* W2    = (const float*)d_in[4];
  const float* b2    = (const float*)d_in[5];
  const float* alpha = (const float*)d_in[6];
  float* out = (float*)d_out;

  // workspace: dinv (32KB) | P1 bf16 8192x256 (4MB) | HH bf16 8192x256 (4MB) | Q bf16 8192x1024 (16MB)
  char* ws = (char*)d_ws;
  float*  dinv = (float*)ws;
  bf16_t* P1 = (bf16_t*)(ws + (32 * 1024));
  bf16_t* HH = (bf16_t*)(ws + (32 * 1024) + (size_t)NN * HID * 2);
  bf16_t* Q  = (bf16_t*)(ws + (32 * 1024) + (size_t)2 * NN * HID * 2);

  dim3 blk(256, 1, 1);
  gcn_degree_kernel<<<dim3(NN, 1, 1), blk, 0, stream>>>(adj, dinv, NN);

  // P1 = dinv ⊙ (x @ W1)                              M=8192 N=256 K=1024
  gcn_wmma_gemm<<<dim3(HID / TILE_N, NN / TILE_M, 1), blk, 0, stream>>>(
      x, nullptr, 0, IN, W1, nullptr, 0, HID, NN, HID, IN, 0,
      dinv, nullptr, nullptr, 0, nullptr, nullptr, P1, nullptr, HID);

  // HH = relu(dinv ⊙ (Abin @ P1 + P1) + b1)           M=8192 N=256 K=8192
  gcn_wmma_gemm<<<dim3(HID / TILE_N, NN / TILE_M, 1), blk, 0, stream>>>(
      adj, nullptr, 2, NN, nullptr, P1, 1, HID, NN, HID, NN, 1,
      dinv, b1, P1, HID, nullptr, nullptr, HH, nullptr, HID);

  // Q = dinv ⊙ (HH @ W2)                              M=8192 N=1024 K=256
  gcn_wmma_gemm<<<dim3(IN / TILE_N, NN / TILE_M, 1), blk, 0, stream>>>(
      nullptr, HH, 1, HID, W2, nullptr, 0, IN, NN, IN, HID, 0,
      dinv, nullptr, nullptr, 0, nullptr, nullptr, Q, nullptr, IN);

  // out = x + alpha*(dinv ⊙ (Abin @ Q + Q) + b2)      M=8192 N=1024 K=8192
  gcn_wmma_gemm<<<dim3(IN / TILE_N, NN / TILE_M, 1), blk, 0, stream>>>(
      adj, nullptr, 2, NN, nullptr, Q, 1, IN, NN, IN, NN, 2,
      dinv, b2, Q, IN, x, alpha, nullptr, out, IN);
}
